// MultiHeadAttention_68547678044247
// MI455X (gfx1250) — matrix operand order repaired and checked by reading, not compiled, not run on hardware
//
#include <hip/hip_runtime.h>

// ---------------- problem constants ----------------
#define B_DIM 2
#define L_DIM 2048
#define C_DIM 256
#define NH    8
#define DHD   32
#define HW    16
#define D_DIM 512                 // DHD * HW : per-head contraction length
#define BH    16                  // NH * B
#define CTX_ELEMS ((size_t)B_DIM * L_DIM * C_DIM * HW)   // 16,777,216 floats
#define SCALE 0.17677669529663687f // 1/sqrt(32)
#define LDSS  40                  // padded bf16 LDS row stride (32+8) = 80B
#define LDSSF 36                  // padded fp32 LDS row stride (32+4) = 144B

typedef __attribute__((ext_vector_type(16))) __bf16 v16bf;
typedef __attribute__((ext_vector_type(8)))  __bf16 bf16x8;
typedef __attribute__((ext_vector_type(8)))  float  v8f;
typedef __attribute__((ext_vector_type(4)))  float  f32x4;

// ---------------- async global->LDS copy (CDNA5 GLOBAL_LOAD_ASYNC_TO_LDS) ----
#if defined(__HIP_DEVICE_COMPILE__) && \
    __has_builtin(__builtin_amdgcn_global_load_async_to_lds_b128)
#define ASYNC_CP 1
#else
#define ASYNC_CP 0
#endif

// Builtin expects pointers to a 16B int vector, addrspace(1) src / addrspace(3) dst.
typedef int v4i __attribute__((__vector_size__(4 * sizeof(int))));
typedef __attribute__((address_space(1))) v4i v4i_g;
typedef __attribute__((address_space(3))) v4i v4i_l;

static __device__ inline void cp16_g2l(void* lds, const void* g) {
#if ASYNC_CP
  __builtin_amdgcn_global_load_async_to_lds_b128((v4i_g*)g, (v4i_l*)lds,
                                                 /*offset=*/0, /*cpol=*/0);
#else
  *(f32x4*)lds = *(const f32x4*)g;   // sync 16B fallback
#endif
}

static __device__ inline void async_wait0() {
#if ASYNC_CP
#if __has_builtin(__builtin_amdgcn_s_wait_asynccnt)
  __builtin_amdgcn_s_wait_asynccnt(0);
#else
  asm volatile("s_wait_asynccnt 0x0" ::: "memory");
#endif
#endif
}

// ---------------- WMMA helpers ----------------
static __device__ inline v8f wmma_bf16(v16bf a, v16bf b, v8f c) {
  return __builtin_amdgcn_wmma_f32_16x16x32_bf16(
      /*neg_a=*/false, a, /*neg_b=*/false, b,
      /*c_mod=*/(short)0, c, /*reuse_a=*/false, /*reuse_b=*/false);
}

// A-matrix fragment (16x32 bf16): lane holds two 8-elem K chunks at p and p+16.
// Caller passes p = base + r*stride + h*8 where r=lane&15, h=lane>>4.
static __device__ inline v16bf make_afrag(const __bf16* p) {
  bf16x8 lo = *(const bf16x8*)p;
  bf16x8 hi = *(const bf16x8*)(p + 16);
  v16bf f;
#pragma unroll
  for (int e = 0; e < 8; ++e) { f[e] = lo[e]; f[e + 8] = hi[e]; }
  return f;
}

// Same A fragment but sourced from an fp32 slab (convert at build time).
static __device__ inline v16bf make_afrag_f32(const float* p) {
  f32x4 a = *(const f32x4*)p,        b_ = *(const f32x4*)(p + 4);
  f32x4 c = *(const f32x4*)(p + 16), d  = *(const f32x4*)(p + 20);
  v16bf f;
#pragma unroll
  for (int e = 0; e < 4; ++e) {
    f[e]      = (__bf16)a[e];
    f[e + 4]  = (__bf16)b_[e];
    f[e + 8]  = (__bf16)c[e];
    f[e + 12] = (__bf16)d[e];
  }
  return f;
}

// B-matrix fragment (32x16 bf16): lane holds two 8-elem K chunks at p and p+16.
// Caller passes p = base + n*stride + h*8 where n=lane&15, h=lane>>4.
static __device__ inline v16bf make_bfrag(const __bf16* p) {
  bf16x8 lo = *(const bf16x8*)p;
  bf16x8 hi = *(const bf16x8*)(p + 16);
  v16bf f;
#pragma unroll
  for (int e = 0; e < 8; ++e) { f[e] = lo[e]; f[e + 8] = hi[e]; }
  return f;
}

// ---------------- kernel 1: 1x1-conv projections ----------------
// grid (B*L, 3); block 256. One WG computes W(256x256) x X(256x16) for one (b,l).
__global__ __launch_bounds__(256) void proj_kernel(
    const float* __restrict__ q_in, const float* __restrict__ k_in,
    const float* __restrict__ v_in,
    const float* __restrict__ Wq, const float* __restrict__ Wk,
    const float* __restrict__ Wv,
    __bf16* __restrict__ qws, __bf16* __restrict__ kws,
    __bf16* __restrict__ vt) {
  __shared__ __attribute__((aligned(16))) __bf16 ldsXt[HW * C_DIM]; // [hw][c]

  const int bl   = blockIdx.x;
  const int b    = bl / L_DIM;
  const int l    = bl % L_DIM;
  const int proj = blockIdx.y;
  const float* src = (proj == 0) ? q_in : (proj == 1) ? k_in : v_in;
  const float* Wm  = (proj == 0) ? Wq   : (proj == 1) ? Wk   : Wv;
  const float* X   = src + (size_t)bl * (C_DIM * HW);

  // Stage X^T (bf16) into LDS: ldsXt[hw*256 + c]. Coalesced 64B reads per thread.
  {
    const int c = threadIdx.x;                     // 0..255
    const f32x4* xp = (const f32x4*)(X + c * HW);
    f32x4 x0 = xp[0], x1 = xp[1], x2 = xp[2], x3 = xp[3];
#pragma unroll
    for (int e = 0; e < 4; ++e) {
      ldsXt[(e +  0) * C_DIM + c] = (__bf16)x0[e];
      ldsXt[(e +  4) * C_DIM + c] = (__bf16)x1[e];
      ldsXt[(e +  8) * C_DIM + c] = (__bf16)x2[e];
      ldsXt[(e + 12) * C_DIM + c] = (__bf16)x3[e];
    }
  }
  __syncthreads();

  const int lane = threadIdx.x & 31;
  const int wv   = threadIdx.x >> 5;     // 8 waves
  const int r    = lane & 15;
  const int h    = lane >> 4;
  const int m0   = wv * 32;              // each wave: 2 M-tiles (m0, m0+16)

  v8f acc0, acc1;
#pragma unroll
  for (int e = 0; e < 8; ++e) { acc0[e] = 0.f; acc1[e] = 0.f; }

#pragma unroll
  for (int k0 = 0; k0 < C_DIM; k0 += 32) {
    v16bf bf = make_bfrag(&ldsXt[r * C_DIM + k0 + h * 8]);
    // A fragments from W (fp32 global -> bf16). 32B-aligned vector loads.
    const float* pa0 = Wm + (m0 + r) * C_DIM + k0 + h * 8;
    const float* pa1 = pa0 + 16 * C_DIM;
    v16bf a0 = make_afrag_f32(pa0);
    v16bf a1 = make_afrag_f32(pa1);
    acc0 = wmma_bf16(a0, bf, acc0);
    acc1 = wmma_bf16(a1, bf, acc1);
  }

  // Scatter outputs. Q/K: bf16 [hb][l][d]; V: transposed bf16 [hb][d][l].
  const int hw = r;
#pragma unroll
  for (int g = 0; g < 8; ++g) {
    const int o0 = m0 + h * 8 + g;
    const int o1 = o0 + 16;
    const float y0 = acc0[g], y1 = acc1[g];
    {
      const int head = o0 >> 5, dh = o0 & 31, hb = head * B_DIM + b;
      if (proj < 2) {
        __bf16* dst = (proj == 0) ? qws : kws;
        dst[((size_t)hb * L_DIM + l) * D_DIM + dh * HW + hw] = (__bf16)y0;
      } else {
        vt[((size_t)hb * D_DIM + dh * HW + hw) * L_DIM + l] = (__bf16)y0;
      }
    }
    {
      const int head = o1 >> 5, dh = o1 & 31, hb = head * B_DIM + b;
      if (proj < 2) {
        __bf16* dst = (proj == 0) ? qws : kws;
        dst[((size_t)hb * L_DIM + l) * D_DIM + dh * HW + hw] = (__bf16)y1;
      } else {
        vt[((size_t)hb * D_DIM + dh * HW + hw) * L_DIM + l] = (__bf16)y1;
      }
    }
  }
}

// ---------------- kernel 2: scores S = scale * Q K^T ----------------
// grid (L/128, L/128, BH); block 256 (8 waves).
// Double-buffered 128x32 bf16 slabs, async-DMA staged (ASYNCcnt).
__global__ __launch_bounds__(256) void score_kernel(
    const __bf16* __restrict__ qws, const __bf16* __restrict__ kws,
    float* __restrict__ attn) {
  __shared__ __attribute__((aligned(16))) __bf16 ldsA[2][128 * LDSS];
  __shared__ __attribute__((aligned(16))) __bf16 ldsB[2][128 * LDSS];

  const int bh = blockIdx.z;
  const int i0 = blockIdx.x * 128;
  const int j0 = blockIdx.y * 128;
  const __bf16* Q = qws + (size_t)bh * L_DIM * D_DIM;
  const __bf16* K = kws + (size_t)bh * L_DIM * D_DIM;

  const int lane = threadIdx.x & 31;
  const int wv   = threadIdx.x >> 5;
  const int r    = lane & 15;
  const int h    = lane >> 4;
  const int srow = threadIdx.x >> 1;        // staging: 0..127
  const int sseg = (threadIdx.x & 1) * 16;  // 0 / 16 (bf16 elems)

  v8f acc[8];
#pragma unroll
  for (int t = 0; t < 8; ++t)
#pragma unroll
    for (int e = 0; e < 8; ++e) acc[t][e] = 0.f;

  const size_t arow = (size_t)(i0 + srow) * D_DIM + sseg;
  const size_t brow = (size_t)(j0 + srow) * D_DIM + sseg;
  const int    lrow = srow * LDSS + sseg;

  // prologue: stage slab 0
  cp16_g2l(&ldsA[0][lrow],     Q + arow);
  cp16_g2l(&ldsA[0][lrow + 8], Q + arow + 8);
  cp16_g2l(&ldsB[0][lrow],     K + brow);
  cp16_g2l(&ldsB[0][lrow + 8], K + brow + 8);

  const int NIT = D_DIM / 32;   // 16
  for (int k = 0; k < NIT; ++k) {
    async_wait0();
    __syncthreads();            // slab k resident; slab k^1 free
    if (k + 1 < NIT) {
      const int nb = (k + 1) & 1;
      const size_t ko = (size_t)(k + 1) * 32;
      cp16_g2l(&ldsA[nb][lrow],     Q + arow + ko);
      cp16_g2l(&ldsA[nb][lrow + 8], Q + arow + ko + 8);
      cp16_g2l(&ldsB[nb][lrow],     K + brow + ko);
      cp16_g2l(&ldsB[nb][lrow + 8], K + brow + ko + 8);
    }
    const int cb = k & 1;
    v16bf a = make_afrag(&ldsA[cb][(wv * 16 + r) * LDSS + h * 8]);
#pragma unroll
    for (int t = 0; t < 8; ++t) {
      v16bf bb = make_bfrag(&ldsB[cb][(t * 16 + r) * LDSS + h * 8]);
      acc[t] = wmma_bf16(a, bb, acc[t]);
    }
  }

  float* S = attn + (size_t)bh * L_DIM * L_DIM;
#pragma unroll
  for (int t = 0; t < 8; ++t) {
    const int j = j0 + t * 16 + r;
#pragma unroll
    for (int g = 0; g < 8; ++g) {
      const int i = i0 + wv * 16 + h * 8 + g;
      S[(size_t)i * L_DIM + j] = acc[t][g] * SCALE;
    }
  }
}

// ---------------- kernel 3: row softmax, in-place ----------------
// grid BH*L; block 256. 8 elements/thread.
__global__ __launch_bounds__(256) void softmax_kernel(float* __restrict__ attn) {
  __shared__ float sbuf[8];
  float* p = attn + (size_t)blockIdx.x * L_DIM;
  const int t = threadIdx.x, lane = t & 31, wv = t >> 5;

  float v[8];
  float m = -3.402823466e38f;
#pragma unroll
  for (int k = 0; k < 8; ++k) { v[k] = p[t + k * 256]; m = fmaxf(m, v[k]); }
#pragma unroll
  for (int off = 16; off > 0; off >>= 1) m = fmaxf(m, __shfl_xor(m, off, 32));
  if (lane == 0) sbuf[wv] = m;
  __syncthreads();
  m = sbuf[0];
#pragma unroll
  for (int k = 1; k < 8; ++k) m = fmaxf(m, sbuf[k]);
  __syncthreads();

  float s = 0.f;
#pragma unroll
  for (int k = 0; k < 8; ++k) { v[k] = __expf(v[k] - m); s += v[k]; }
#pragma unroll
  for (int off = 16; off > 0; off >>= 1) s += __shfl_xor(s, off, 32);
  if (lane == 0) sbuf[wv] = s;
  __syncthreads();
  s = sbuf[0];
#pragma unroll
  for (int k = 1; k < 8; ++k) s += sbuf[k];
  const float inv = 1.0f / s;
#pragma unroll
  for (int k = 0; k < 8; ++k) p[t + k * 256] = v[k] * inv;
}

// ---------------- kernel 4: context = attn @ V ----------------
// grid (L/128, D/128, BH); block 256.
// attn slab DMA'd raw fp32 into LDS, converted to bf16 at fragment build.
__global__ __launch_bounds__(256) void context_kernel(
    const float* __restrict__ attn, const __bf16* __restrict__ vt,
    float* __restrict__ ctx) {
  __shared__ __attribute__((aligned(16))) float  ldsAf[2][128 * LDSSF];
  __shared__ __attribute__((aligned(16))) __bf16 ldsB[2][128 * LDSS];

  const int bh = blockIdx.z;
  const int i0 = blockIdx.x * 128;
  const int n0 = blockIdx.y * 128;
  const float*  P  = attn + (size_t)bh * L_DIM * L_DIM;
  const __bf16* Vt = vt   + (size_t)bh * D_DIM * L_DIM;

  const int lane = threadIdx.x & 31;
  const int wv   = threadIdx.x >> 5;
  const int r    = lane & 15;
  const int h    = lane >> 4;
  const int srow = threadIdx.x >> 1;
  const int sseg = (threadIdx.x & 1) * 16;

  v8f acc[8];
#pragma unroll
  for (int t = 0; t < 8; ++t)
#pragma unroll
    for (int e = 0; e < 8; ++e) acc[t][e] = 0.f;

  const size_t arow = (size_t)(i0 + srow) * L_DIM + sseg;  // fp32 elems
  const size_t brow = (size_t)(n0 + srow) * L_DIM + sseg;  // bf16 elems
  const int    larw = srow * LDSSF + sseg;
  const int    lbrw = srow * LDSS + sseg;

  // prologue: stage slab 0
  cp16_g2l(&ldsAf[0][larw],      P + arow);
  cp16_g2l(&ldsAf[0][larw + 4],  P + arow + 4);
  cp16_g2l(&ldsAf[0][larw + 8],  P + arow + 8);
  cp16_g2l(&ldsAf[0][larw + 12], P + arow + 12);
  cp16_g2l(&ldsB[0][lbrw],       Vt + brow);
  cp16_g2l(&ldsB[0][lbrw + 8],   Vt + brow + 8);

  const int NIT = L_DIM / 32;   // 64
  for (int k = 0; k < NIT; ++k) {
    async_wait0();
    __syncthreads();
    if (k + 1 < NIT) {
      const int nb = (k + 1) & 1;
      const size_t ko = (size_t)(k + 1) * 32;
      cp16_g2l(&ldsAf[nb][larw],      P + arow + ko);
      cp16_g2l(&ldsAf[nb][larw + 4],  P + arow + ko + 4);
      cp16_g2l(&ldsAf[nb][larw + 8],  P + arow + ko + 8);
      cp16_g2l(&ldsAf[nb][larw + 12], P + arow + ko + 12);
      cp16_g2l(&ldsB[nb][lbrw],       Vt + brow + ko);
      cp16_g2l(&ldsB[nb][lbrw + 8],   Vt + brow + ko + 8);
    }
    const int cb = k & 1;
    v16bf a = make_afrag_f32(&ldsAf[cb][(wv * 16 + r) * LDSSF + h * 8]);
#pragma unroll
    for (int t = 0; t < 8; ++t) {
      v16bf bb = make_bfrag(&ldsB[cb][(t * 16 + r) * LDSS + h * 8]);
      acc[t] = wmma_bf16(a, bb, acc[t]);
    }
  }

  // Scatter back into (B, L, C, H, W) with c = head*32 + dh, hw = d & 15.
  const int headv = bh / B_DIM;
  const int bb    = bh % B_DIM;
#pragma unroll
  for (int t = 0; t < 8; ++t) {
    const int d  = n0 + t * 16 + r;
    const int dh = d >> 4, hw = d & 15;
    const int cc = headv * DHD + dh;
#pragma unroll
    for (int g = 0; g < 8; ++g) {
      const int i = i0 + wv * 16 + h * 8 + g;
      ctx[(((size_t)bb * L_DIM + i) * C_DIM + cc) * HW + hw] = acc[t][g];
    }
  }
}

// ---------------- host launcher ----------------
extern "C" void kernel_launch(void* const* d_in, const int* in_sizes, int n_in,
                              void* d_out, int out_size, void* d_ws,
                              size_t ws_size, hipStream_t stream) {
  (void)in_sizes; (void)n_in; (void)out_size; (void)ws_size;
  const float* q_in = (const float*)d_in[0];
  const float* k_in = (const float*)d_in[1];
  const float* v_in = (const float*)d_in[2];
  const float* Wq   = (const float*)d_in[3];
  const float* Wk   = (const float*)d_in[4];
  const float* Wv   = (const float*)d_in[5];

  float* ctx  = (float*)d_out;
  float* attn = (float*)d_out + CTX_ELEMS;

  const size_t QK_BYTES = (size_t)BH * L_DIM * D_DIM * 2; // 33.5 MB each
  char* ws = (char*)d_ws;
  __bf16* qws = (__bf16*)(ws);
  __bf16* kws = (__bf16*)(ws + QK_BYTES);
  __bf16* vt  = (__bf16*)(ws + 2 * QK_BYTES);

  proj_kernel<<<dim3(B_DIM * L_DIM, 3), 256, 0, stream>>>(
      q_in, k_in, v_in, Wq, Wk, Wv, qws, kws, vt);
  score_kernel<<<dim3(L_DIM / 128, L_DIM / 128, BH), 256, 0, stream>>>(
      qws, kws, attn);
  softmax_kernel<<<BH * L_DIM, 256, 0, stream>>>(attn);
  context_kernel<<<dim3(L_DIM / 128, D_DIM / 128, BH), 256, 0, stream>>>(
      attn, vt, ctx);
}
